// CA_82978768159300
// MI455X (gfx1250) — matrix operand order repaired and checked by reading, not compiled, not run on hardware
//
#include <hip/hip_runtime.h>
#include <hip/hip_bf16.h>
#include <hip/hip_fp16.h>

typedef __attribute__((ext_vector_type(16))) _Float16 v16h;
typedef __attribute__((ext_vector_type(8)))  float    v8f;

#define B_ 8
#define T_ 8
#define IN_CH 11
#define CIN 32
#define HID 64
#define KBANK 128
#define FEAT 256
#define HW 64
#define NIMG 64            // B_*T_
#define NITER 10
#define BN_N 262144.0f     // B*T*H*W
#define EPS 1e-5f

union FragH { v16h h; uint4 q[2]; };

// ---------------------------------------------------------------------------
// Selection weights: w = softmax(tf @ sel_w + sel_b), mixed biases b1/b2.
// ---------------------------------------------------------------------------
__global__ __launch_bounds__(128) void k_sel(
    const float* __restrict__ tf,
    const float* __restrict__ sw1, const float* __restrict__ sb1, const float* __restrict__ bb1,
    const float* __restrict__ sw2, const float* __restrict__ sb2, const float* __restrict__ bb2,
    float* __restrict__ w1, float* __restrict__ w2,
    float* __restrict__ b1, float* __restrict__ b2)
{
    __shared__ float lg[B_ * KBANK];
    __shared__ float mxs[B_], sms[B_];
    const int k = threadIdx.x;   // 0..127
    for (int pass = 0; pass < 2; ++pass) {
        const float* sw = pass ? sw2 : sw1;
        const float* sb = pass ? sb2 : sb1;
        float* wout     = pass ? w2  : w1;
        for (int b = 0; b < B_; ++b) {
            float acc = sb[k];
            for (int f = 0; f < FEAT; ++f) acc += tf[b * FEAT + f] * sw[f * KBANK + k];
            lg[b * KBANK + k] = acc;
        }
        __syncthreads();
        if (k < B_) {
            float m = -3.0e38f;
            for (int j = 0; j < KBANK; ++j) m = fmaxf(m, lg[k * KBANK + j]);
            float s = 0.f;
            for (int j = 0; j < KBANK; ++j) s += __expf(lg[k * KBANK + j] - m);
            mxs[k] = m; sms[k] = s;
        }
        __syncthreads();
        for (int b = 0; b < B_; ++b)
            wout[b * KBANK + k] = __expf(lg[b * KBANK + k] - mxs[b]) / sms[b];
        __syncthreads();
    }
    if (k < HID)
        for (int b = 0; b < B_; ++b) {
            float a = 0.f;
            for (int j = 0; j < KBANK; ++j) a += w1[b * KBANK + j] * bb1[j * HID + k];
            b1[b * HID + k] = a;
        }
    if (k < CIN)
        for (int b = 0; b < B_; ++b) {
            float a = 0.f;
            for (int j = 0; j < KBANK; ++j) a += w2[b * KBANK + j] * bb2[j * CIN + k];
            b2[b * CIN + k] = a;
        }
}

// ---------------------------------------------------------------------------
// Mix bank1 -> k1h[b][tap][o=64][i=32] (f16)
// ---------------------------------------------------------------------------
__global__ __launch_bounds__(256) void k_mix1(
    const float* __restrict__ w1, const float* __restrict__ bank,
    _Float16* __restrict__ kh)
{
    int t = blockIdx.x * 256 + threadIdx.x;     // 8*9*64*32 = 147456
    if (t >= 147456) return;
    int i = t & 31, o = (t >> 5) & 63, bt = t >> 11, tap = bt % 9, b = bt / 9;
    float acc = 0.f;
    for (int k = 0; k < KBANK; ++k)
        acc += w1[b * KBANK + k] * bank[((size_t)k * 2048 + o * 32 + i) * 9 + tap];
    kh[t] = (_Float16)acc;
}

// Mix bank2 -> k2h[b][tap][o=32][i=64] (f16)
__global__ __launch_bounds__(256) void k_mix2(
    const float* __restrict__ w2, const float* __restrict__ bank,
    _Float16* __restrict__ kh)
{
    int t = blockIdx.x * 256 + threadIdx.x;     // 147456
    if (t >= 147456) return;
    int i = t & 63, o = (t >> 6) & 31, bt = t >> 11, tap = bt % 9, b = bt / 9;
    float acc = 0.f;
    for (int k = 0; k < KBANK; ++k)
        acc += w2[b * KBANK + k] * bank[((size_t)k * 2048 + o * 64 + i) * 9 + tap];
    kh[t] = (_Float16)acc;
}

// ---------------------------------------------------------------------------
// x0: first 11 channels = infer_inputs, latent = 1.0 (f16). Zero BN accum.
// ---------------------------------------------------------------------------
__global__ __launch_bounds__(256) void k_init(
    const float* __restrict__ inp, _Float16* __restrict__ x0,
    float* __restrict__ bn_acc)
{
    if (blockIdx.x == 0 && threadIdx.x < 2 * HID) bn_acc[threadIdx.x] = 0.f;
    int t = blockIdx.x * 256 + threadIdx.x;     // 64*32*4096
    if (t >= NIMG * CIN * 4096) return;
    int px = t & 4095, ch = (t >> 12) & 31, img = t >> 17;
    _Float16 v = (_Float16)1.0f;
    if (ch < IN_CH) v = (_Float16)inp[((size_t)img * IN_CH + ch) * 4096 + px];
    x0[t] = v;
}

// ---------------------------------------------------------------------------
// conv1: h = conv3x3(x, k1) + b1  (WMMA), accumulate BN sum/sumsq.
// Block: one output row (64px) of one image; 8 waves.
// Wave = 16 px x 32 out-channels (2 M-fragments) so each B fragment feeds
// two WMMAs (halves LDS read traffic vs. one M-tile per wave).
// ---------------------------------------------------------------------------
__global__ __launch_bounds__(256) void k_conv1(
    const _Float16* __restrict__ xin,   // [64][32][64][64]
    const _Float16* __restrict__ k1h,   // [8][9][64][32]
    const float*    __restrict__ b1,    // [8][64]
    _Float16*       __restrict__ hout,  // [64][64][64][64]
    float*          __restrict__ bn_acc)// sum[64], sumsq[64]
{
    __shared__ alignas(16) _Float16 tile[3 * 66 * 32];
    __shared__ float red[2 * HID];
    const int yrow = blockIdx.x;
    const int img  = blockIdx.y;
    const int b    = img >> 3;
    const int tid  = threadIdx.x;

    if (tid < 2 * HID) red[tid] = 0.f;
    if (tid < 192) {                     // zero halo columns x=-1, x=64
        int r = tid / 64, which = (tid >> 5) & 1, ch = tid & 31;
        tile[(r * 66 + (which ? 65 : 0)) * 32 + ch] = (_Float16)0.f;
    }
    // warm L0 with this image's weight block while staging the input tile
    if (tid < 72)
        __builtin_prefetch(k1h + (size_t)b * 9 * HID * CIN + tid * 256, 0, 0);
    for (int c = tid; c < 768; c += 256) {   // 3 rows * 32ch * 8 groups
        int r = c >> 8, rem = c & 255, ch = rem >> 3, x0 = (rem & 7) * 8;
        int yy = yrow + r - 1;
        uint4 data = make_uint4(0u, 0u, 0u, 0u);
        if (yy >= 0 && yy < HW)
            data = *(const uint4*)(xin + (((size_t)img * CIN + ch) * HW + yy) * HW + x0);
        const _Float16* hp = (const _Float16*)&data;
        #pragma unroll
        for (int j = 0; j < 8; ++j)
            tile[(r * 66 + x0 + 1 + j) * 32 + ch] = hp[j];
    }
    __syncthreads();

    const int wave  = tid >> 5, lane = tid & 31, lm = lane & 15;
    const bool hi   = lane >= 16;
    const int xi    = (wave & 3) * 16 + lm;   // pixel within row
    const int mbase = (wave >> 2) * 32;       // 0 or 32
    const int kb    = hi ? 8 : 0;
    const int cb    = hi ? 16 : 0;
    const int chb   = hi ? 8 : 0;

    v8f c0, c1;
    #pragma unroll
    for (int i = 0; i < 8; ++i) {
        c0[i] = b1[b * HID + mbase + chb + i];
        c1[i] = b1[b * HID + mbase + 16 + chb + i];
    }
    const _Float16* wp = k1h + (size_t)b * 9 * HID * CIN;
    #pragma unroll
    for (int tap = 0; tap < 9; ++tap) {
        const int r = tap / 3, dx = tap % 3;
        FragH bf;
        const _Float16* bp = &tile[(r * 66 + xi + dx) * 32 + kb];
        bf.q[0] = *(const uint4*)bp;
        bf.q[1] = *(const uint4*)(bp + 16);
        FragH a0, a1;
        const _Float16* p0 = wp + ((size_t)tap * HID + mbase + lm) * CIN;
        const _Float16* p1 = wp + ((size_t)tap * HID + mbase + 16 + lm) * CIN;
        a0.q[0] = *(const uint4*)(p0 + kb);  a0.q[1] = *(const uint4*)(p0 + 16 + kb);
        a1.q[0] = *(const uint4*)(p1 + kb);  a1.q[1] = *(const uint4*)(p1 + 16 + kb);
        c0 = __builtin_amdgcn_wmma_f32_16x16x32_f16(
                 false, a0.h, false, bf.h, (short)0, c0, false, false);
        c1 = __builtin_amdgcn_wmma_f32_16x16x32_f16(
                 false, a1.h, false, bf.h, (short)0, c1, false, false);
    }
    #pragma unroll
    for (int f = 0; f < 2; ++f) {
        #pragma unroll
        for (int i = 0; i < 8; ++i) {
            const int ch = mbase + f * 16 + chb + i;
            const float v = (f == 0) ? c0[i] : c1[i];
            hout[(((size_t)img * HID + ch) * HW + yrow) * HW + xi] = (_Float16)v;
            atomicAdd(&red[ch], v);
            atomicAdd(&red[HID + ch], v * v);
        }
    }
    __syncthreads();
    if (tid < 2 * HID) unsafeAtomicAdd(&bn_acc[tid], red[tid]);
}

// ---------------------------------------------------------------------------
// BN finalize: per-channel scale/shift; re-zero accumulators.
// ---------------------------------------------------------------------------
__global__ __launch_bounds__(64) void k_bnfin(
    float* __restrict__ acc, const float* __restrict__ gamma,
    const float* __restrict__ beta, float* __restrict__ ss)
{
    const int ch = threadIdx.x;
    const float mean = acc[ch] / BN_N;
    const float var  = acc[HID + ch] / BN_N - mean * mean;
    const float inv  = rsqrtf(var + EPS);
    const float sc   = gamma[ch] * inv;
    ss[ch]       = sc;
    ss[HID + ch] = beta[ch] - mean * sc;
    acc[ch] = 0.f; acc[HID + ch] = 0.f;
}

// ---------------------------------------------------------------------------
// conv2: y = conv3x3(lrelu(bn(h)), k2) + b2 ; softmax(11) / tanh ; emit log-sm.
// Block: one output row of one image; 4 waves, each wave = 16 px, all 32 ch.
// ---------------------------------------------------------------------------
__global__ __launch_bounds__(128) void k_conv2(
    const _Float16* __restrict__ hbuf,  // [64][64][64][64]
    const float*    __restrict__ ss,    // scale[64], shift[64]
    const _Float16* __restrict__ k2h,   // [8][9][32][64]
    const float*    __restrict__ b2,    // [8][32]
    _Float16*       __restrict__ xnext, // [64][32][64][64]
    float*          __restrict__ out,   // [8][8][10][11][64][64]
    int iter)
{
    __shared__ alignas(16) _Float16 tile[3 * 66 * 64];
    const int yrow = blockIdx.x;
    const int img  = blockIdx.y;
    const int b    = img >> 3;
    const int tid  = threadIdx.x;

    for (int z = tid; z < 384; z += 128) {   // halo columns
        int r = z / 128, which = (z >> 6) & 1, ch = z & 63;
        tile[(r * 66 + (which ? 65 : 0)) * 64 + ch] = (_Float16)0.f;
    }
    if (tid < 128)                            // warm L0 with weight block
        __builtin_prefetch(k2h + (size_t)b * 9 * CIN * HID + tid * 144, 0, 0);
    for (int c = tid; c < 1536; c += 128) {  // 3 rows * 64ch * 8 groups
        int r = c / 512, rem = c % 512, ch = rem >> 3, x0 = (rem & 7) * 8;
        int yy = yrow + r - 1;
        const bool inb = (yy >= 0 && yy < HW);
        uint4 data = make_uint4(0u, 0u, 0u, 0u);
        if (inb)
            data = *(const uint4*)(hbuf + (((size_t)img * HID + ch) * HW + yy) * HW + x0);
        const float sc = ss[ch], sh = ss[HID + ch];
        const _Float16* hp = (const _Float16*)&data;
        #pragma unroll
        for (int j = 0; j < 8; ++j) {
            float v = sc * (float)hp[j] + sh;
            v = v >= 0.f ? v : 0.5f * v;
            tile[(r * 66 + x0 + 1 + j) * 64 + ch] = inb ? (_Float16)v : (_Float16)0.f;
        }
    }
    __syncthreads();

    const int wave = tid >> 5, lane = tid & 31, lm = lane & 15;
    const bool hi  = lane >= 16;
    const int xi   = wave * 16 + lm;
    const int kb   = hi ? 8 : 0;
    const int cb   = hi ? 16 : 0;

    v8f c0, c1;
    #pragma unroll
    for (int i = 0; i < 8; ++i) {
        c0[i] = b2[b * CIN + i + (hi ? 8 : 0)];
        c1[i] = b2[b * CIN + 16 + i + (hi ? 8 : 0)];
    }
    const _Float16* wbase = k2h + (size_t)b * 9 * CIN * HID;
    #pragma unroll
    for (int tap = 0; tap < 9; ++tap) {
        const int r = tap / 3, dx = tap % 3;
        #pragma unroll
        for (int ks = 0; ks < 2; ++ks) {
            FragH bf;
            const _Float16* bp = &tile[(r * 66 + xi + dx) * 64 + ks * 32 + kb];
            bf.q[0] = *(const uint4*)bp;
            bf.q[1] = *(const uint4*)(bp + 16);
            FragH a0, a1;
            const _Float16* p0 = wbase + ((size_t)tap * CIN + lm) * HID + ks * 32;
            const _Float16* p1 = wbase + ((size_t)tap * CIN + 16 + lm) * HID + ks * 32;
            a0.q[0] = *(const uint4*)(p0 + kb);  a0.q[1] = *(const uint4*)(p0 + 16 + kb);
            a1.q[0] = *(const uint4*)(p1 + kb);  a1.q[1] = *(const uint4*)(p1 + 16 + kb);
            c0 = __builtin_amdgcn_wmma_f32_16x16x32_f16(
                     false, a0.h, false, bf.h, (short)0, c0, false, false);
            c1 = __builtin_amdgcn_wmma_f32_16x16x32_f16(
                     false, a1.h, false, bf.h, (short)0, c1, false, false);
        }
    }

    // lane holds ch = chb..chb+7 (frag0) and 16+chb.. (frag1), chb = hi?8:0
    const int chb = hi ? 8 : 0;
    float mx = -3.0e38f;
    #pragma unroll
    for (int i = 0; i < 8; ++i)
        if (chb + i < IN_CH) mx = fmaxf(mx, c0[i]);
    mx = fmaxf(mx, __shfl_xor(mx, 16, 32));
    float se = 0.f;
    #pragma unroll
    for (int i = 0; i < 8; ++i)
        if (chb + i < IN_CH) se += __expf(c0[i] - mx);
    se += __shfl_xor(se, 16, 32);
    const float lse = __logf(se);
    const float inv = 1.0f / se;

    const size_t xoff = (size_t)img * CIN * 4096 + (size_t)yrow * HW + xi;
    const size_t ooff = ((size_t)(img * NITER + iter)) * IN_CH * 4096 + (size_t)yrow * HW + xi;
    #pragma unroll
    for (int f = 0; f < 2; ++f) {
        #pragma unroll
        for (int i = 0; i < 8; ++i) {
            const int ch = f * 16 + chb + i;
            const float v = (f == 0) ? c0[i] : c1[i];
            _Float16 xv;
            if (ch < IN_CH) {
                xv = (_Float16)(__expf(v - mx) * inv);
                out[ooff + (size_t)ch * 4096] = (v - mx) - lse;
            } else {
                xv = (_Float16)tanhf(v);
            }
            xnext[xoff + (size_t)ch * 4096] = xv;
        }
    }
}

// ---------------------------------------------------------------------------
extern "C" void kernel_launch(void* const* d_in, const int* in_sizes, int n_in,
                              void* d_out, int out_size, void* d_ws, size_t ws_size,
                              hipStream_t stream) {
    (void)in_sizes; (void)n_in; (void)out_size; (void)ws_size;
    const float* tf    = (const float*)d_in[0];
    const float* infer = (const float*)d_in[1];
    const float* bank1 = (const float*)d_in[2];
    const float* sw1   = (const float*)d_in[3];
    const float* sb1   = (const float*)d_in[4];
    const float* bb1   = (const float*)d_in[5];
    const float* bank2 = (const float*)d_in[6];
    const float* sw2   = (const float*)d_in[7];
    const float* sb2   = (const float*)d_in[8];
    const float* bb2   = (const float*)d_in[9];
    const float* gamma = (const float*)d_in[10];
    const float* beta  = (const float*)d_in[11];
    float* out = (float*)d_out;

    char* ws = (char*)d_ws;
    size_t off = 0;
    auto alloc = [&](size_t bytes) { size_t r = off; off += (bytes + 255) & ~(size_t)255; return r; };
    _Float16* xa  = (_Float16*)(ws + alloc((size_t)NIMG * CIN * 4096 * 2));   // 16 MB
    _Float16* xb  = (_Float16*)(ws + alloc((size_t)NIMG * CIN * 4096 * 2));   // 16 MB
    _Float16* hb  = (_Float16*)(ws + alloc((size_t)NIMG * HID * 4096 * 2));   // 32 MB
    _Float16* k1h = (_Float16*)(ws + alloc((size_t)B_ * 9 * HID * CIN * 2));
    _Float16* k2h = (_Float16*)(ws + alloc((size_t)B_ * 9 * CIN * HID * 2));
    float* w1 = (float*)(ws + alloc(B_ * KBANK * 4));
    float* w2 = (float*)(ws + alloc(B_ * KBANK * 4));
    float* b1 = (float*)(ws + alloc(B_ * HID * 4));
    float* b2 = (float*)(ws + alloc(B_ * CIN * 4));
    float* bnacc = (float*)(ws + alloc(2 * HID * 4));
    float* ssb   = (float*)(ws + alloc(2 * HID * 4));

    k_sel<<<1, 128, 0, stream>>>(tf, sw1, sb1, bb1, sw2, sb2, bb2, w1, w2, b1, b2);
    k_mix1<<<576, 256, 0, stream>>>(w1, bank1, k1h);
    k_mix2<<<576, 256, 0, stream>>>(w2, bank2, k2h);
    k_init<<<(NIMG * CIN * 4096 + 255) / 256, 256, 0, stream>>>(infer, xa, bnacc);

    _Float16* xc = xa;
    _Float16* xn = xb;
    for (int it = 0; it < NITER; ++it) {
        k_conv1<<<dim3(HW, NIMG), 256, 0, stream>>>(xc, k1h, b1, hb, bnacc);
        k_bnfin<<<1, 64, 0, stream>>>(bnacc, gamma, beta, ssb);
        k_conv2<<<dim3(HW, NIMG), 128, 0, stream>>>(hb, ssb, k2h, b2, xn, out, it);
        _Float16* tmp = xc; xc = xn; xn = tmp;
    }
}
